// MemoryModule_13606456394201
// MI455X (gfx1250) — compile-verified, matrix operand order checked by reading, not run on hardware
//
#include <hip/hip_runtime.h>
#include <hip/hip_bf16.h>
#include <math.h>
#include <stdint.h>

typedef __attribute__((ext_vector_type(16))) _Float16 v16h;
typedef __attribute__((ext_vector_type(8)))  _Float16 v8h;
typedef __attribute__((ext_vector_type(8)))  float    v8f;
typedef __attribute__((ext_vector_type(4)))  int      v4i;

constexpr int E_    = 131072;
constexpr int NN    = 32768;
constexpr int DM    = 256;
constexpr int DE    = 128;
constexpr int TD    = 64;
constexpr int HIDN  = 512;
constexpr int MSGN  = 256;
constexpr int INDIM = 2 * DM + DE + TD;   // 704
constexpr int LSTR  = 40;                 // LDS row stride in halves (80B, conflict-free)

// ---------------------------------------------------------------------------
// CDNA5 async global->LDS staging (GLOBAL_LOAD_ASYNC_TO_LDS_B128, ASYNCcnt).
// Builtin signature (from compiler diagnostic): takes v4i pointers in
// addrspace(1)/(3) plus immediate offset + cpol.
// Falls back to a synchronous VGPR->LDS copy if the builtin is unavailable.
// ---------------------------------------------------------------------------
#if __has_builtin(__builtin_amdgcn_global_load_async_to_lds_b128) && \
    __has_builtin(__builtin_amdgcn_s_wait_asynccnt)
#define HAVE_ASYNC_LDS 1
#else
#define HAVE_ASYNC_LDS 0
#endif

#define AS_GLOBAL __attribute__((address_space(1)))
#define AS_LOCAL  __attribute__((address_space(3)))

__device__ __forceinline__ void async_cp16(const void* g, void* lds) {
#if HAVE_ASYNC_LDS
    // Generic LDS pointer: low 32 bits == LDS byte offset (flat aperture rule).
    __builtin_amdgcn_global_load_async_to_lds_b128(
        (AS_GLOBAL v4i*)(uintptr_t)g,
        (AS_LOCAL  v4i*)(uint32_t)(uintptr_t)lds,
        0, 0);
#else
    *(v8h*)lds = *(const v8h*)g;
#endif
}

template <int N>
__device__ __forceinline__ void wait_async() {
#if HAVE_ASYNC_LDS
    __builtin_amdgcn_s_wait_asynccnt(N);
#endif
}

// ---------------------------------------------------------------------------
// WMMA fragment load from an LDS tile (16 rows x 32 halves, row stride LSTR).
// Per ISA 16-bit A/B layout: lanes 0-15 hold row=lane, K {0..7,16..23};
// lanes 16-31 hold row=lane-16, K {8..15,24..31}.
// ---------------------------------------------------------------------------
__device__ __forceinline__ v16h ld_frag(const _Float16* tile, int lane) {
    const int row = lane & 15;
    const int kb  = (lane >> 4) << 3;           // 0 or 8
    const _Float16* p = tile + row * LSTR;
    v8h lo = *(const v8h*)(p + kb);             // K = kb .. kb+7
    v8h hi = *(const v8h*)(p + kb + 16);        // K = kb+16 .. kb+23
    v16h f;
#pragma unroll
    for (int i = 0; i < 8; ++i) { f[i] = lo[i]; f[i + 8] = hi[i]; }
    return f;
}

// ---------------------------------------------------------------------------
// GEMM1 (fused): h = relu(x @ w1^T + b1), x built on the fly in LDS.
// Block: 256 threads (8 waves). BM=32, BN=512 (full N -> x streamed once).
// B tile staged via async global->LDS, launched BEFORE the VALU-heavy
// feature build so the fetch hides under the cos/convert work.
// ---------------------------------------------------------------------------
__global__ __launch_bounds__(256)
void gemm1_fused(const float* __restrict__ relt,
                 const float* __restrict__ srcm,
                 const float* __restrict__ dstm,
                 const float* __restrict__ edgef,
                 const float* __restrict__ timew,
                 const float* __restrict__ timeb,
                 const _Float16* __restrict__ w1h,
                 const float* __restrict__ b1,
                 _Float16* __restrict__ Hout) {
    __shared__ alignas(16) _Float16 sA[32 * LSTR];
    __shared__ alignas(16) _Float16 sB[512 * LSTR];

    const int t    = threadIdx.x;
    const int lane = t & 31;
    const int wid  = t >> 5;
    const long long m0 = (long long)blockIdx.x * 32;

    v8f acc[2][4] = {};

    for (int k0 = 0; k0 < INDIM; k0 += 32) {
        // ---- issue async B stage (w1 tile 512x32): 8 x b128 per wave ----
        {
            const int rb = t >> 2, c = t & 3;
            const int kk = k0 + c * 8;
#pragma unroll
            for (int i = 0; i < 8; ++i) {
                const int n = rb + i * 64;
                async_cp16(w1h + (long long)n * INDIM + kk, &sB[n * LSTR + c * 8]);
            }
        }
        // ---- stage A (x tile 32x32): build features in f16 (VALU) ----
        if (t < 128) {
            const int row = t >> 2, c = t & 3;
            const long long e = m0 + row;
            const int kk = k0 + c * 8;
            _Float16* da = &sA[row * LSTR + c * 8];
            if (kk < 2 * DM) {
                const float* s = (kk < DM) ? (srcm + e * DM + kk)
                                           : (dstm + e * DM + (kk - DM));
#pragma unroll
                for (int i = 0; i < 8; ++i) da[i] = (_Float16)s[i];
            } else if (kk < 2 * DM + DE) {
                const float* s = edgef + e * DE + (kk - 2 * DM);
#pragma unroll
                for (int i = 0; i < 8; ++i) da[i] = (_Float16)s[i];
            } else {
                const float rt = relt[e];
                const int j0 = kk - (2 * DM + DE);
#pragma unroll
                for (int i = 0; i < 8; ++i)
                    da[i] = (_Float16)__cosf(rt * timew[j0 + i] + timeb[j0 + i]);
            }
        }
        wait_async<0>();
        __syncthreads();

        v16h af[2];
#pragma unroll
        for (int mt = 0; mt < 2; ++mt) af[mt] = ld_frag(&sA[mt * 16 * LSTR], lane);
#pragma unroll
        for (int nt = 0; nt < 4; ++nt) {
            v16h bf = ld_frag(&sB[(wid * 64 + nt * 16) * LSTR], lane);
#pragma unroll
            for (int mt = 0; mt < 2; ++mt)
                acc[mt][nt] = __builtin_amdgcn_wmma_f32_16x16x32_f16(
                    false, af[mt], false, bf, (short)0, acc[mt][nt], false, false);
        }
        __syncthreads();
    }

    // ---- epilogue: bias + relu, store f16 h ----
    const int g = lane >> 4;
#pragma unroll
    for (int mt = 0; mt < 2; ++mt)
#pragma unroll
        for (int nt = 0; nt < 4; ++nt) {
            const int n = wid * 64 + nt * 16 + (lane & 15);
            const float bv = b1[n];
#pragma unroll
            for (int r = 0; r < 8; ++r) {
                const long long m = m0 + mt * 16 + r + 8 * g;
                float v = acc[mt][nt][r] + bv;
                v = fmaxf(v, 0.0f);
                Hout[m * HIDN + n] = (_Float16)v;
            }
        }
}

// ---------------------------------------------------------------------------
// Generic WMMA GEMM: C(f32, MxN) = A(f16, MxK) @ W(f16, NxK)^T + bias.
// Block 256 threads: BM=64, BN=128, waves 2x4, each wave 32x32 (2x2 tiles).
// Double-buffered async global->LDS staging: loads for step k+1 fly while
// WMMAs chew on step k (3 async b128 per wave per step; s_wait_asynccnt 3).
// ---------------------------------------------------------------------------
__global__ __launch_bounds__(256)
void gemm_wmma_f32(const _Float16* __restrict__ A,
                   const _Float16* __restrict__ W,
                   const float* __restrict__ bias,
                   float* __restrict__ C,
                   int M, int N, int K) {
    __shared__ alignas(16) _Float16 sA[2][64 * LSTR];
    __shared__ alignas(16) _Float16 sB[2][128 * LSTR];

    const int t    = threadIdx.x;
    const int lane = t & 31;
    const int wid  = t >> 5;
    const int waveM = wid >> 2, waveN = wid & 3;
    const long long m0 = (long long)blockIdx.x * 64;
    const long long n0 = (long long)blockIdx.y * 128;

    const int rowA = t >> 2;       // 0..63
    const int c    = t & 3;        // k-chunk
    const int rb   = t >> 2;       // B row base

    auto issue = [&](int buf, int k0) {
        const int kk = k0 + c * 8;
        async_cp16(A + (m0 + rowA) * K + kk, &sA[buf][rowA * LSTR + c * 8]);
#pragma unroll
        for (int i = 0; i < 2; ++i) {
            const int n = rb + i * 64;
            async_cp16(W + (n0 + n) * K + kk, &sB[buf][n * LSTR + c * 8]);
        }
    };

    v8f acc[2][2] = {};
    const int steps = K / 32;

    issue(0, 0);
    for (int it = 0; it < steps; ++it) {
        const int cur = it & 1;
        if (it + 1 < steps) {
            issue(cur ^ 1, (it + 1) * 32);
            wait_async<3>();           // newest 3 may stay in flight
        } else {
            wait_async<0>();
        }
        __syncthreads();

        v16h af[2];
#pragma unroll
        for (int mt = 0; mt < 2; ++mt)
            af[mt] = ld_frag(&sA[cur][(waveM * 32 + mt * 16) * LSTR], lane);
#pragma unroll
        for (int nt = 0; nt < 2; ++nt) {
            v16h bf = ld_frag(&sB[cur][(waveN * 32 + nt * 16) * LSTR], lane);
#pragma unroll
            for (int mt = 0; mt < 2; ++mt)
                acc[mt][nt] = __builtin_amdgcn_wmma_f32_16x16x32_f16(
                    false, af[mt], false, bf, (short)0, acc[mt][nt], false, false);
        }
        __syncthreads();
    }

    const int g = lane >> 4;
#pragma unroll
    for (int mt = 0; mt < 2; ++mt)
#pragma unroll
        for (int nt = 0; nt < 2; ++nt) {
            const long long n = n0 + waveN * 32 + nt * 16 + (lane & 15);
            const float bv = bias[n];
#pragma unroll
            for (int r = 0; r < 8; ++r) {
                const long long m = m0 + waveM * 32 + mt * 16 + r + 8 * g;
                C[m * N + n] = acc[mt][nt][r] + bv;
            }
        }
}

// ------------------------------- helpers -----------------------------------
__global__ void k_f32_to_f16(const float* __restrict__ src,
                             _Float16* __restrict__ dst, long long n) {
    long long i = (long long)blockIdx.x * blockDim.x + threadIdx.x;
    if (i < n) dst[i] = (_Float16)src[i];
}

__global__ void k_zero_i32(int* p, long long n) {
    long long i = (long long)blockIdx.x * blockDim.x + threadIdx.x;
    if (i < n) p[i] = 0;
}

__global__ void k_zero_f32(float* p, long long n) {
    long long i = (long long)blockIdx.x * blockDim.x + threadIdx.x;
    if (i < n) p[i] = 0.0f;
}

__global__ void k_fill_ids(long long* p, long long n, long long v) {
    long long i = (long long)blockIdx.x * blockDim.x + threadIdx.x;
    if (i < n) p[i] = v;
}

__global__ void k_hist(const long long* __restrict__ dst_ids, int* __restrict__ counts) {
    long long e = (long long)blockIdx.x * blockDim.x + threadIdx.x;
    if (e < E_) atomicAdd(&counts[(int)dst_ids[e]], 1);
}

// Single-block prefix scan of presence flags over NN=32768 -> rank per node.
__global__ __launch_bounds__(1024)
void k_scan_ranks(const int* __restrict__ counts, int* __restrict__ rank) {
    __shared__ int part[1024];
    const int t = threadIdx.x;
    const int base = t * 32;
    int loc[32];
    int s = 0;
#pragma unroll
    for (int i = 0; i < 32; ++i) { loc[i] = s; s += (counts[base + i] > 0) ? 1 : 0; }
    part[t] = s;
    __syncthreads();
    for (int off = 1; off < 1024; off <<= 1) {
        int v = (t >= off) ? part[t - off] : 0;
        __syncthreads();
        part[t] += v;
        __syncthreads();
    }
    const int excl = part[t] - s;
#pragma unroll
    for (int i = 0; i < 32; ++i) rank[base + i] = excl + loc[i];
}

__global__ void k_scatter_ids(const int* __restrict__ counts,
                              const int* __restrict__ rank,
                              long long* __restrict__ uniq) {
    int v = blockIdx.x * blockDim.x + threadIdx.x;
    if (v < NN && counts[v] > 0) uniq[rank[v]] = (long long)v;
}

// One block per edge: atomic scatter-add of its msg row into sums[rank].
__global__ __launch_bounds__(256)
void k_scatter_add(const long long* __restrict__ dst_ids,
                   const int* __restrict__ rank,
                   const float* __restrict__ msg,
                   float* __restrict__ sums) {
    const long long e = blockIdx.x;
    const int rk = rank[(int)dst_ids[e]];
    const int j = threadIdx.x;
    atomicAdd(&sums[(long long)rk * MSGN + j], msg[e * MSGN + j]);
}

__global__ void k_agg_finalize(const float* __restrict__ sums,
                               const long long* __restrict__ uniq,
                               const int* __restrict__ counts,
                               _Float16* __restrict__ aggh) {
    long long i = (long long)blockIdx.x * blockDim.x + threadIdx.x;
    if (i >= (long long)NN * MSGN) return;
    const long long r = i >> 8;
    const long long node = uniq[r];
    const int cnum = (node < NN) ? counts[node] : 0;
    aggh[i] = (_Float16)(sums[i] / fmaxf((float)cnum, 1.0f));
}

__device__ __forceinline__ float sigm(float x) { return 1.0f / (1.0f + __expf(-x)); }

__global__ void k_gru_gates(const float* __restrict__ gi,
                            const float* __restrict__ gh,
                            const float* __restrict__ prev,  // dst_memories[:NN]
                            float* __restrict__ out_mem) {
    long long i = (long long)blockIdx.x * blockDim.x + threadIdx.x;
    if (i >= (long long)NN * DM) return;
    const long long row = i >> 8;
    const int j = (int)(i & 255);
    const float* giR = gi + row * (3 * DM);
    const float* ghR = gh + row * (3 * DM);
    const float r = sigm(giR[j] + ghR[j]);
    const float z = sigm(giR[DM + j] + ghR[DM + j]);
    const float n = tanhf(giR[2 * DM + j] + r * ghR[2 * DM + j]);
    out_mem[i] = (1.0f - z) * n + z * prev[i];
}

// ---------------------------------------------------------------------------
extern "C" void kernel_launch(void* const* d_in, const int* in_sizes, int n_in,
                              void* d_out, int out_size, void* d_ws, size_t ws_size,
                              hipStream_t stream) {
    const float*     rel_time = (const float*)d_in[0];
    const float*     srcm     = (const float*)d_in[1];
    const float*     dstm     = (const float*)d_in[2];
    const float*     edgef    = (const float*)d_in[3];
    const long long* dst_ids  = (const long long*)d_in[4];
    const float*     timew    = (const float*)d_in[5];
    const float*     timeb    = (const float*)d_in[6];
    const float*     w1       = (const float*)d_in[7];
    const float*     b1       = (const float*)d_in[8];
    const float*     w2       = (const float*)d_in[9];
    const float*     b2       = (const float*)d_in[10];
    const float*     wih      = (const float*)d_in[11];
    const float*     whh      = (const float*)d_in[12];
    const float*     bih      = (const float*)d_in[13];
    const float*     bhh      = (const float*)d_in[14];

    // ---- workspace carve-out (256B aligned) ----
    char* ws = (char*)d_ws;
    size_t off = 0;
    auto carve = [&](size_t bytes) -> char* {
        off = (off + 255) & ~(size_t)255;
        char* p = ws + off;
        off += bytes;
        return p;
    };
    _Float16* Hh    = (_Float16*)carve((size_t)E_ * HIDN * 2);   // h (f16)
    float*    MSGf  = (float*)   carve((size_t)E_ * MSGN * 4);   // msg (f32)
    float*    sums  = (float*)   carve((size_t)NN * MSGN * 4);
    _Float16* aggh  = (_Float16*)carve((size_t)NN * MSGN * 2);
    _Float16* prevh = (_Float16*)carve((size_t)NN * DM * 2);
    _Float16* w1h   = (_Float16*)carve((size_t)HIDN * INDIM * 2);
    _Float16* w2h   = (_Float16*)carve((size_t)MSGN * HIDN * 2);
    _Float16* wihh  = (_Float16*)carve((size_t)3 * DM * MSGN * 2);
    _Float16* whhh  = (_Float16*)carve((size_t)3 * DM * DM * 2);
    int*      counts= (int*)     carve((size_t)NN * 4);
    int*      rank  = (int*)     carve((size_t)NN * 4);
    // aliases over dead regions:
    float* gi = (float*)Hh;    // 32768x768 f32 (96MB)  <= Hh region (128MB)
    float* gh = (float*)MSGf;  // 32768x768 f32 (96MB)  <= MSG region (128MB)

    long long* uniq_out = (long long*)d_out;             // 32768 int64
    float*     mem_out  = (float*)d_out + 2 * NN;        // 32768x256 f32

    const int TB = 256;
    auto blocks = [&](long long n) { return (unsigned)((n + TB - 1) / TB); };

    // 1) weight / prev conversions to f16
    k_f32_to_f16<<<blocks((long long)HIDN * INDIM), TB, 0, stream>>>(w1, w1h, (long long)HIDN * INDIM);
    k_f32_to_f16<<<blocks((long long)MSGN * HIDN), TB, 0, stream>>>(w2, w2h, (long long)MSGN * HIDN);
    k_f32_to_f16<<<blocks((long long)3 * DM * MSGN), TB, 0, stream>>>(wih, wihh, (long long)3 * DM * MSGN);
    k_f32_to_f16<<<blocks((long long)3 * DM * DM), TB, 0, stream>>>(whh, whhh, (long long)3 * DM * DM);
    k_f32_to_f16<<<blocks((long long)NN * DM), TB, 0, stream>>>(dstm, prevh, (long long)NN * DM);

    // 2) zero accumulators, pre-fill unique ids with fill_value = NN
    k_zero_i32<<<blocks(NN), TB, 0, stream>>>(counts, NN);
    k_zero_f32<<<blocks((long long)NN * MSGN), TB, 0, stream>>>(sums, (long long)NN * MSGN);
    k_fill_ids<<<blocks(NN), TB, 0, stream>>>(uniq_out, NN, (long long)NN);

    // 3) GEMM1 fused: h = relu(x@w1^T + b1), x built in LDS
    gemm1_fused<<<dim3(E_ / 32), TB, 0, stream>>>(rel_time, srcm, dstm, edgef,
                                                  timew, timeb, w1h, b1, Hh);

    // 4) GEMM2: msg = h@w2^T + b2
    gemm_wmma_f32<<<dim3(E_ / 64, MSGN / 128), TB, 0, stream>>>(Hh, w2h, b2, MSGf,
                                                                E_, MSGN, HIDN);

    // 5) unique + segment mean
    k_hist<<<blocks(E_), TB, 0, stream>>>(dst_ids, counts);
    k_scan_ranks<<<1, 1024, 0, stream>>>(counts, rank);
    k_scatter_ids<<<blocks(NN), TB, 0, stream>>>(counts, rank, uniq_out);
    k_scatter_add<<<dim3(E_), TB, 0, stream>>>(dst_ids, rank, MSGf, sums);
    k_agg_finalize<<<blocks((long long)NN * MSGN), TB, 0, stream>>>(sums, uniq_out, counts, aggh);

    // 6) GRU GEMMs: gi = agg@w_ih^T + b_ih ; gh = prev@w_hh^T + b_hh
    gemm_wmma_f32<<<dim3(NN / 64, (3 * DM) / 128), TB, 0, stream>>>(aggh, wihh, bih, gi,
                                                                    NN, 3 * DM, MSGN);
    gemm_wmma_f32<<<dim3(NN / 64, (3 * DM) / 128), TB, 0, stream>>>(prevh, whhh, bhh, gh,
                                                                    NN, 3 * DM, DM);

    // 7) gates -> memories
    k_gru_gates<<<blocks((long long)NN * DM), TB, 0, stream>>>(gi, gh, dstm, mem_out);
}
